// PointNetPolylineEncoder_8959301780155
// MI455X (gfx1250) — matrix operand order repaired and checked by reading, not compiled, not run on hardware
//
#include <hip/hip_runtime.h>

typedef __attribute__((ext_vector_type(16))) _Float16 v16h;
typedef __attribute__((ext_vector_type(8)))  _Float16 v8h;
typedef __attribute__((ext_vector_type(8)))  float    v8f;

#define BN_EPS 1e-5f

// Problem sizes
#define NPOLY 16384   // B*N = 16*1024
#define PPTS  32
#define CIN   9
#define H     64
#define OUTC  128

// wpack layout (offsets in _Float16 elements). Each fragment = 512 halves
// (32 lanes x 16 halves), stored so a lane loads its v16h at
//   wpack + base + (frag*512 + lane*16)
#define WS_PRE 0        // 1 kchunk x 4 ntiles  = 4  frags
#define WS_M1  2048     // 4 kchunks x 4 ntiles = 16 frags
#define WS_M2  10240    // 2 x 4 = 8 frags
#define WS_O1  14336    // 2 x 4 = 8 frags
#define WS_O2  18432    // 2 x 8 = 16 frags
#define WPACK_HALVES 26624

// byte offsets in d_ws
#define WSB_BN    53248      // 384 f32: [layer*128 + ch]=scale, [layer*128+64+ch]=shift
#define WSB_VALID 55296      // 16384 f32 (0/1 per polyline)
#define WSB_FEAT  131072     // 16384*64 f16 pooled features

static __device__ __forceinline__ v16h cat8(v8h lo, v8h hi) {
  return __builtin_shufflevector(lo, hi, 0,1,2,3,4,5,6,7,8,9,10,11,12,13,14,15);
}

static __device__ __forceinline__ v8f wmma16(v16h a, v16h b, v8f c) {
  return __builtin_amdgcn_wmma_f32_16x16x32_f16(false, a, false, b, (short)0, c, false, false);
}

// ---------------- prep: pack weights into WMMA B-fragment layout, fold BN ----
static __device__ __forceinline__ void pack_one(_Float16* dst, const float* W,
                                                int e, int NT, int Kact, int Ntot) {
  int frag = e >> 9;          // /512
  int within = e & 511;
  int lane = within >> 4;
  int i = within & 15;
  int nt = frag % NT;
  int c  = frag / NT;
  int k, n;
  if (lane < 16) { k = c*32 + (i < 8 ? i : i + 8);     n = nt*16 + lane;      }
  else           { k = c*32 + 8 + (i < 8 ? i : i + 8); n = nt*16 + lane - 16; }
  float v = (k < Kact) ? W[k*Ntot + n] : 0.0f;
  dst[e] = (_Float16)v;
}

__global__ void k_prep(const float* preW, const float* m1W, const float* m2W,
                       const float* o1W, const float* o2W,
                       const float* pb,  const float* pg,  const float* pbe,
                       const float* prm, const float* prv,
                       const float* m1b, const float* m1g, const float* m1be,
                       const float* m1rm,const float* m1rv,
                       const float* m2b, const float* m2g, const float* m2be,
                       const float* m2rm,const float* m2rv,
                       _Float16* wpack, float* bnc) {
  int i = blockIdx.x * blockDim.x + threadIdx.x;
  if (i < 2048)                 pack_one(wpack + WS_PRE, preW, i,            4,   9,  64);
  else if (i < 10240)           pack_one(wpack + WS_M1,  m1W,  i - 2048,     4, 128,  64);
  else if (i < 14336)           pack_one(wpack + WS_M2,  m2W,  i - 10240,    4,  64,  64);
  else if (i < 18432)           pack_one(wpack + WS_O1,  o1W,  i - 14336,    4,  64,  64);
  else if (i < 26624)           pack_one(wpack + WS_O2,  o2W,  i - 18432,    8,  64, 128);
  else if (i < 26624 + 192) {
    int j = i - 26624;
    int layer = j >> 6, ch = j & 63;
    const float *b, *g, *be, *rm, *rv;
    if (layer == 0)      { b = pb;  g = pg;  be = pbe;  rm = prm;  rv = prv;  }
    else if (layer == 1) { b = m1b; g = m1g; be = m1be; rm = m1rm; rv = m1rv; }
    else                 { b = m2b; g = m2g; be = m2be; rm = m2rm; rv = m2rv; }
    float s = g[ch] * rsqrtf(rv[ch] + BN_EPS);
    bnc[layer*128 + ch]      = s;
    bnc[layer*128 + 64 + ch] = (b[ch] - rm[ch]) * s + be[ch];
  }
}

// ---------------- encoder: one wave per polyline -----------------------------
__global__ __launch_bounds__(32) void k_encoder(
    const float* __restrict__ X, const unsigned char* __restrict__ Msk,
    const _Float16* __restrict__ wpack, const float* __restrict__ bnc,
    float* __restrict__ validF, _Float16* __restrict__ featH) {
  const int poly  = blockIdx.x;
  const int lane  = threadIdx.x;
  const int hgrp  = lane >> 4;     // which 16-lane half
  const int lcol  = lane & 15;

  __shared__ _Float16 fbuf[PPTS * H]  __attribute__((aligned(16)));  // f after pre
  __shared__ _Float16 h1buf[PPTS * H] __attribute__((aligned(16)));  // after m1
  __shared__ _Float16 pooledh[H]      __attribute__((aligned(16)));

  // Per-lane row-validity. Tile t, this lane's D rows = t*16 + 8*hgrp + r.
  float mrow[2][8];
  unsigned int anym = 0;
  #pragma unroll
  for (int t = 0; t < 2; ++t) {
    const unsigned char* mp = Msk + (size_t)poly*PPTS + t*16 + hgrp*8;
    uint2 mb = *(const uint2*)mp;
    anym |= mb.x | mb.y;
    #pragma unroll
    for (int r = 0; r < 4; ++r) {
      mrow[t][r]     = ((mb.x >> (8*r)) & 0xffu) ? 1.0f : 0.0f;
      mrow[t][4 + r] = ((mb.y >> (8*r)) & 0xffu) ? 1.0f : 0.0f;
    }
  }

  // ---- A-fragments for pre layer: x[32,9] padded to K=32 -------------------
  v16h a1[2];
  #pragma unroll
  for (int t = 0; t < 2; ++t) {
    v16h a = {};
    if (lane < 16) {
      const float* xp = X + ((size_t)poly*PPTS + t*16 + lane) * CIN;
      #pragma unroll
      for (int j = 0; j < 8; ++j) a[j] = (_Float16)xp[j];   // K=0..7
    } else {
      const float* xp = X + ((size_t)poly*PPTS + t*16 + (lane - 16)) * CIN;
      a[0] = (_Float16)xp[8];                               // K=8
    }
    a1[t] = a;
  }

  // ---- Stage A: pre MLP + BN + ReLU + mask; running col-max for pooling ----
  float cmax[4] = {0.f, 0.f, 0.f, 0.f};
  #pragma unroll
  for (int t = 0; t < 2; ++t) {
    #pragma unroll
    for (int nt = 0; nt < 4; ++nt) {
      v16h bf = *(const v16h*)(wpack + WS_PRE + (nt*512 + lane*16));
      v8f acc = {};
      acc = wmma16(a1[t], bf, acc);
      int col = nt*16 + lcol;
      float sc = bnc[col], sh = bnc[64 + col];
      #pragma unroll
      for (int r = 0; r < 8; ++r) {
        float v = fmaxf(acc[r]*sc + sh, 0.0f) * mrow[t][r];
        cmax[nt] = fmaxf(cmax[nt], v);
        fbuf[(t*16 + hgrp*8 + r)*H + col] = (_Float16)v;
      }
    }
  }
  __syncthreads();
  #pragma unroll
  for (int nt = 0; nt < 4; ++nt) {
    float pm = fmaxf(cmax[nt], __shfl_xor(cmax[nt], 16, 32));
    if (lane < 16) pooledh[nt*16 + lane] = (_Float16)pm;
  }
  __syncthreads();

  const int so = hgrp ? 8 : 0;   // K-half offset inside A fragments

  // ---- Stage B: m1 (K=128: k<64 from fbuf rows, k>=64 pooled broadcast) ----
  #pragma unroll
  for (int t = 0; t < 2; ++t) {
    v16h af[4];
    const _Float16* rp = fbuf + (t*16 + lcol)*H;
    #pragma unroll
    for (int c = 0; c < 2; ++c)
      af[c] = cat8(*(const v8h*)(rp + c*32 + so), *(const v8h*)(rp + c*32 + 16 + so));
    #pragma unroll
    for (int c = 2; c < 4; ++c)
      af[c] = cat8(*(const v8h*)(pooledh + (c-2)*32 + so),
                   *(const v8h*)(pooledh + (c-2)*32 + 16 + so));
    #pragma unroll
    for (int nt = 0; nt < 4; ++nt) {
      v8f acc = {};
      #pragma unroll
      for (int c = 0; c < 4; ++c) {
        v16h bf = *(const v16h*)(wpack + WS_M1 + ((c*4 + nt)*512 + lane*16));
        acc = wmma16(af[c], bf, acc);
      }
      int col = nt*16 + lcol;
      float sc = bnc[128 + col], sh = bnc[128 + 64 + col];
      #pragma unroll
      for (int r = 0; r < 8; ++r) {
        float v = fmaxf(acc[r]*sc + sh, 0.0f);
        h1buf[(t*16 + hgrp*8 + r)*H + col] = (_Float16)v;
      }
    }
  }
  __syncthreads();

  // ---- Stage C: m2 + BN + ReLU + mask; masked max -> feat ------------------
  float fmx[4] = {0.f, 0.f, 0.f, 0.f};
  #pragma unroll
  for (int t = 0; t < 2; ++t) {
    v16h af[2];
    const _Float16* rp = h1buf + (t*16 + lcol)*H;
    #pragma unroll
    for (int c = 0; c < 2; ++c)
      af[c] = cat8(*(const v8h*)(rp + c*32 + so), *(const v8h*)(rp + c*32 + 16 + so));
    #pragma unroll
    for (int nt = 0; nt < 4; ++nt) {
      v8f acc = {};
      #pragma unroll
      for (int c = 0; c < 2; ++c) {
        v16h bf = *(const v16h*)(wpack + WS_M2 + ((c*4 + nt)*512 + lane*16));
        acc = wmma16(af[c], bf, acc);
      }
      int col = nt*16 + lcol;
      float sc = bnc[256 + col], sh = bnc[256 + 64 + col];
      #pragma unroll
      for (int r = 0; r < 8; ++r) {
        float v = fmaxf(acc[r]*sc + sh, 0.0f) * mrow[t][r];
        fmx[nt] = fmaxf(fmx[nt], v);
      }
    }
  }
  #pragma unroll
  for (int nt = 0; nt < 4; ++nt) {
    float pm = fmaxf(fmx[nt], __shfl_xor(fmx[nt], 16, 32));
    if (lane < 16) featH[(size_t)poly*H + nt*16 + lane] = (_Float16)pm;
  }
  bool valid = (__ballot(anym != 0u) != 0ull);
  if (lane == 0) validF[poly] = valid ? 1.0f : 0.0f;
}

// ---------------- out MLP: one wave per 16 polylines -------------------------
__global__ __launch_bounds__(32) void k_out(
    const _Float16* __restrict__ featH, const float* __restrict__ validF,
    const _Float16* __restrict__ wpack,
    const float* __restrict__ b1, const float* __restrict__ b2,
    float* __restrict__ out) {
  const int base = blockIdx.x * 16;
  const int lane = threadIdx.x;
  const int hgrp = lane >> 4;
  const int lcol = lane & 15;
  const int so = hgrp ? 8 : 0;

  __shared__ _Float16 hbuf[16 * H] __attribute__((aligned(16)));

  // A fragments: feat[16,64] (K=64 -> 2 chunks)
  v16h af[2];
  {
    const _Float16* rp = featH + ((size_t)base + lcol) * H;
    #pragma unroll
    for (int c = 0; c < 2; ++c)
      af[c] = cat8(*(const v8h*)(rp + c*32 + so), *(const v8h*)(rp + c*32 + 16 + so));
  }
  // GEMM1: feat @ out1_w + b1, ReLU
  #pragma unroll
  for (int nt = 0; nt < 4; ++nt) {
    v8f acc = {};
    #pragma unroll
    for (int c = 0; c < 2; ++c) {
      v16h bf = *(const v16h*)(wpack + WS_O1 + ((c*4 + nt)*512 + lane*16));
      acc = wmma16(af[c], bf, acc);
    }
    int col = nt*16 + lcol;
    float bb = b1[col];
    #pragma unroll
    for (int r = 0; r < 8; ++r)
      hbuf[(hgrp*8 + r)*H + col] = (_Float16)fmaxf(acc[r] + bb, 0.0f);
  }
  __syncthreads();

  v16h ag[2];
  {
    const _Float16* rp = hbuf + lcol * H;
    #pragma unroll
    for (int c = 0; c < 2; ++c)
      ag[c] = cat8(*(const v8h*)(rp + c*32 + so), *(const v8h*)(rp + c*32 + 16 + so));
  }
  float varr[8];
  #pragma unroll
  for (int r = 0; r < 8; ++r) varr[r] = validF[base + hgrp*8 + r];

  // GEMM2: h @ out2_w + b2, validity mask, f32 store
  #pragma unroll
  for (int nt = 0; nt < 8; ++nt) {
    v8f acc = {};
    #pragma unroll
    for (int c = 0; c < 2; ++c) {
      v16h bf = *(const v16h*)(wpack + WS_O2 + ((c*8 + nt)*512 + lane*16));
      acc = wmma16(ag[c], bf, acc);
    }
    int col = nt*16 + lcol;
    float bb = b2[col];
    #pragma unroll
    for (int r = 0; r < 8; ++r)
      out[((size_t)base + hgrp*8 + r)*OUTC + col] = (acc[r] + bb) * varr[r];
  }
}

extern "C" void kernel_launch(void* const* d_in, const int* in_sizes, int n_in,
                              void* d_out, int out_size, void* d_ws, size_t ws_size,
                              hipStream_t stream) {
  (void)in_sizes; (void)n_in; (void)out_size; (void)ws_size;
  const float* X = (const float*)d_in[0];
  const unsigned char* Msk = (const unsigned char*)d_in[1];

  _Float16* wpack  = (_Float16*)d_ws;
  float*    bnc    = (float*)((char*)d_ws + WSB_BN);
  float*    validF = (float*)((char*)d_ws + WSB_VALID);
  _Float16* featH  = (_Float16*)((char*)d_ws + WSB_FEAT);

  // 1) pack weights + fold BN
  int prep_items = WPACK_HALVES + 192;
  k_prep<<<(prep_items + 255) / 256, 256, 0, stream>>>(
      (const float*)d_in[2],  (const float*)d_in[8],  (const float*)d_in[14],
      (const float*)d_in[20], (const float*)d_in[22],
      (const float*)d_in[3],  (const float*)d_in[4],  (const float*)d_in[5],
      (const float*)d_in[6],  (const float*)d_in[7],
      (const float*)d_in[9],  (const float*)d_in[10], (const float*)d_in[11],
      (const float*)d_in[12], (const float*)d_in[13],
      (const float*)d_in[15], (const float*)d_in[16], (const float*)d_in[17],
      (const float*)d_in[18], (const float*)d_in[19],
      wpack, bnc);

  // 2) per-polyline point MLPs -> feat + valid
  k_encoder<<<NPOLY, 32, 0, stream>>>(X, Msk, wpack, bnc, validF, featH);

  // 3) out MLP over feat (16 polylines per wave)
  k_out<<<NPOLY / 16, 32, 0, stream>>>(featH, validF, wpack,
                                       (const float*)d_in[21], (const float*)d_in[23],
                                       (float*)d_out);
}
